// ROIHead_MambaLite_51737176048190
// MI455X (gfx1250) — matrix operand order repaired and checked by reading, not compiled, not run on hardware
//
#include <hip/hip_runtime.h>
#include <hip/hip_bf16.h>
#include <stdint.h>

// ---------------- problem constants ----------------
#define N_ROI  1024
#define C_DIM  256
#define L_DIM  64
#define DI     512      // 2*C
#define NLAY   4
#define S_DIM  16
#define R_DIM  16
#define K_CONV 4
#define H_DIM  512
#define EPS_LN 1e-5f

typedef unsigned short u16;
typedef __attribute__((ext_vector_type(16))) __bf16 v16bf;
typedef __attribute__((ext_vector_type(8)))  __bf16 v8bf;
typedef __attribute__((ext_vector_type(8)))  float  v8f;
typedef __attribute__((ext_vector_type(4)))  unsigned int v4u;
typedef __attribute__((ext_vector_type(8)))  int v8i;
typedef __attribute__((ext_vector_type(4)))  int v4i;

#if defined(__gfx1250__) && __has_builtin(__builtin_amdgcn_tensor_load_to_lds) && \
    __has_builtin(__builtin_amdgcn_s_wait_tensorcnt)
#define HAVE_TDM 1
#else
#define HAVE_TDM 0
#endif

// ---------------- bf16 helpers (bit-level, RNE) ----------------
__device__ __forceinline__ float bf2f(u16 h) {
  unsigned x = ((unsigned)h) << 16;
  return __builtin_bit_cast(float, x);
}
__device__ __forceinline__ u16 f2bf(float f) {
  unsigned x = __builtin_bit_cast(unsigned, f);
  unsigned r = (x + 0x7FFFu + ((x >> 16) & 1u)) >> 16;
  return (u16)r;
}
__device__ __forceinline__ float silu_f(float v) { return v / (1.f + __expf(-v)); }
__device__ __forceinline__ float softplus_f(float v) {
  return (v > 20.f) ? v : log1pf(__expf(v));
}

// ---------------- WMMA fragment helpers ----------------
// A-matrix 16x32 bf16, contiguous k order used by this code (wave32): lane L: row = L&15, kh = L>>4,
//   elem i -> K = kh*16 + i
__device__ __forceinline__ v16bf load_a_frag(const u16* rowk0, int kh) {
  return *reinterpret_cast<const v16bf*>(rowk0 + kh * 16);
}
// B-matrix 32x16 bf16, same contiguous k order as A here: lane L: col = L&15, kh = L>>4, elem i -> K = kh*16+i.
// Weights stored transposed (row = out col, K contiguous) -> one 32B load.
__device__ __forceinline__ v16bf load_b_frag(const u16* wrow_k0, int kh) {
  return *reinterpret_cast<const v16bf*>(wrow_k0 + kh * 16);
}
__device__ __forceinline__ v8f wmma_bf(v16bf a, v16bf b, v8f c) {
  return __builtin_amdgcn_wmma_f32_16x16x32_bf16(false, a, false, b,
                                                 (short)0, c, false, false);
}

// ---------------- TDM: 1-D 32KB tile, global -> LDS ----------------
// D# per CDNA5 ISA: group0 = {count/flags, lds_addr,
// global_addr[31:0], global_addr[56:32]|type=2}; group1 packs data_size=8B,
// tensor_dim0/tile_dim0 = n8 (8-byte units), dim strides = n8.
// Toolchain builtin is the 6-arg form: (g0, g1, g2, g3, g4, cpol).
__device__ __forceinline__ void tdm_load_1d(unsigned lds_off, const void* gptr,
                                            unsigned n8) {
#if HAVE_TDM
  unsigned long long ga = (unsigned long long)(uintptr_t)gptr;
  v4u g0;
  g0[0] = 1u;                                       // count=1, no gather
  g0[1] = lds_off;                                  // LDS byte address
  g0[2] = (unsigned)ga;                             // global_addr[31:0]
  g0[3] = (unsigned)((ga >> 32) & 0x1FFFFFFu) | (2u << 30);  // [56:32]|type=2
  v8i g1;
  g1[0] = (int)(3u << 16);          // workgroup_mask=0, data_size=3 (8B)
  g1[1] = (int)((n8 & 0xFFFFu) << 16);  // tensor_dim0[15:0]
  g1[2] = (int)(((n8 >> 16) & 0xFFFFu) | (1u << 16));  // dim0[31:16], tensor_dim1=1
  g1[3] = (int)((n8 & 0xFFFFu) << 16);  // tile_dim0 = n8
  g1[4] = 0;                            // tile_dim1=0 (unused), tile_dim2=0
  g1[5] = (int)n8;                      // tensor_dim0_stride[31:0]
  g1[6] = (int)((n8 & 0xFFFFu) << 16);  // dim1_stride[15:0], dim0_stride[47:32]=0
  g1[7] = 0;
  v4i g2 = {0, 0, 0, 0};
  v4i g3 = {0, 0, 0, 0};
  v8i g4 = {0, 0, 0, 0, 0, 0, 0, 0};
  __builtin_amdgcn_tensor_load_to_lds(g0, g1, g2, g3, g4, 0);
#else
  (void)lds_off; (void)gptr; (void)n8;
#endif
}

// ---------------- workspace layout (bf16, pre-transposed weights) ----------
#define WS_WIN_OFF   0
#define WS_WIN_ELEMS (NLAY * 1024 * 256)      // w_inT[l][n][k]
#define WS_WOUT_OFF  (WS_WIN_OFF + WS_WIN_ELEMS * 2)
#define WS_WOUT_ELEMS (NLAY * 256 * 512)      // w_outT[l][n][k]
#define WS_XP_OFF    (WS_WOUT_OFF + WS_WOUT_ELEMS * 2)
#define WS_XP_ELEMS  (2 * NLAY * 48 * 512)    // xpT[dir][l][n][k]
#define WS_TOTAL_ELEMS (WS_WIN_ELEMS + WS_WOUT_ELEMS + WS_XP_ELEMS)

__global__ void prep_weights(const float* __restrict__ w_in,
                             const float* __restrict__ w_out,
                             const float* __restrict__ xp_f,
                             const float* __restrict__ xp_b,
                             u16* __restrict__ w_inT,
                             u16* __restrict__ w_outT,
                             u16* __restrict__ xpT) {
  int idx = blockIdx.x * blockDim.x + threadIdx.x;
  if (idx < WS_WIN_ELEMS) {
    int l = idx / (1024 * 256);
    int rem = idx % (1024 * 256);
    int n = rem / 256, k = rem % 256;
    w_inT[idx] = f2bf(w_in[(l * 256 + k) * 1024 + n]);
  } else if (idx < WS_WIN_ELEMS + WS_WOUT_ELEMS) {
    int j = idx - WS_WIN_ELEMS;
    int l = j / (256 * 512);
    int rem = j % (256 * 512);
    int n = rem / 512, k = rem % 512;
    w_outT[j] = f2bf(w_out[(l * 512 + k) * 256 + n]);
  } else if (idx < WS_TOTAL_ELEMS) {
    int j = idx - WS_WIN_ELEMS - WS_WOUT_ELEMS;
    int dir = j / (NLAY * 48 * 512);
    int rem = j % (NLAY * 48 * 512);
    int l = rem / (48 * 512);
    int rem2 = rem % (48 * 512);
    int n = rem2 / 512, k = rem2 % 512;
    const float* src = dir ? xp_b : xp_f;
    xpT[j] = f2bf(src[(l * 512 + k) * 48 + n]);
  }
}

// ---------------- fused per-ROI forward ----------------
struct Params {
  const float* x_flat;
  const float* ln_g;  const float* ln_b;
  const float* cw[2]; const float* cb[2];
  const float* dtw[2]; const float* dtb[2];
  const float* Alog[2]; const float* Dp[2];
  const float* proj_w; const float* proj_b;
  const u16* w_inT; const u16* w_outT; const u16* xpT;
  float* out;
};

// LDS carve (dynamic, 300KB of the 320KB/WGP):
#define OFF_XS 0                    // bf16 [64][256] residual x      32KB
#define OFF_XI 32768                // bf16 [64][512] xi              64KB
#define OFF_SZ (32768 + 65536)      // bf16 [64][512] silu(z)         64KB
#define OFF_UB (32768 + 2*65536)    // bf16 [64][512] conv out (xn)   64KB
#define OFF_YB (32768 + 3*65536)    // bf16 [64][512] y accum / g     64KB
#define OFF_PR (32768 + 4*65536)    // f32  [64][48]  dtr|B|C         12KB
#define SMEM_BYTES (32768 + 4 * 65536 + 12288)
// During GEMM1 the yB region doubles as a 2x32KB TDM staging buffer for w_inT;
// during GEMM3 the xi region doubles as a 2x32KB staging buffer for w_outT.

__global__ void __launch_bounds__(256, 1)
fused_roi_mamba(Params P) {
  extern __shared__ __align__(16) char smem[];
  u16*   xS = (u16*)(smem + OFF_XS);
  u16*   xi = (u16*)(smem + OFF_XI);
  u16*   sz = (u16*)(smem + OFF_SZ);
  u16*   uB = (u16*)(smem + OFF_UB);
  u16*   yB = (u16*)(smem + OFF_YB);
  float* pR = (float*)(smem + OFF_PR);
  u16*   xn = uB;  // xn (64x256 bf16) lives in uB region during GEMM1

  const int roi  = blockIdx.x;
  const int tid  = threadIdx.x;
  const int wave = tid >> 5;
  const int lane = tid & 31;
  const int nl   = lane & 15;
  const int kh   = lane >> 4;

  // ---- load x tile: x[t][c] = x_flat[roi][c*64 + t] ----
  {
    const float* xf = P.x_flat + (size_t)roi * (C_DIM * L_DIM);
    for (int i = tid; i < C_DIM * L_DIM; i += 256) {
      int c = i >> 6, t = i & 63;
      xS[t * C_DIM + c] = f2bf(xf[i]);
    }
  }
  __syncthreads();

  for (int l = 0; l < NLAY; ++l) {
    // ================= LayerNorm -> xn (bf16) =================
    {
      const int row = wave * 8 + ((tid & 31) >> 2);
      const int sub = tid & 3;
      float s1 = 0.f, s2 = 0.f;
      const u16* xr = xS + row * C_DIM;
      for (int j = 0; j < 64; ++j) {
        float v = bf2f(xr[sub * 64 + j]);
        s1 += v; s2 += v * v;
      }
      s1 += __shfl_xor(s1, 1); s2 += __shfl_xor(s2, 1);
      s1 += __shfl_xor(s1, 2); s2 += __shfl_xor(s2, 2);
      float mu = s1 * (1.f / C_DIM);
      float var = s2 * (1.f / C_DIM) - mu * mu;
      float rs = __frsqrt_rn(var + EPS_LN);
      const float* g = P.ln_g + l * C_DIM;
      const float* b = P.ln_b + l * C_DIM;
      for (int j = 0; j < 64; ++j) {
        int c = sub * 64 + j;
        float v = (bf2f(xr[c]) - mu) * rs * g[c] + b[c];
        xn[row * C_DIM + c] = f2bf(v);
      }
    }
    __syncthreads();

    // ======= GEMM1: xz = xn(64x256) @ w_inT[l] (1024 outs), TDM-staged ======
    // 16 chunks of 4 n-tiles (64 rows x 256 K = 32KB), double-buffered in yB.
    {
      const int m = wave & 3;
      const u16* arow = xn + (m * 16 + nl) * C_DIM;
      v16bf afr[8];
#pragma unroll
      for (int kk = 0; kk < 8; ++kk) afr[kk] = load_a_frag(arow + kk * 32, kh);
      const u16* wl = P.w_inT + (size_t)l * 1024 * 256;
      u16* stage = yB;
      if (tid < 32) tdm_load_1d(OFF_YB, wl, 4096);
      for (int c = 0; c < 16; ++c) {
        if (tid < 32) {
          if (c + 1 < 16) {
            tdm_load_1d(OFF_YB + ((c + 1) & 1) * 32768, wl + (c + 1) * 16384, 4096);
#if HAVE_TDM
            __builtin_amdgcn_s_wait_tensorcnt(1);  // chunk c done; c+1 in flight
#endif
          } else {
#if HAVE_TDM
            __builtin_amdgcn_s_wait_tensorcnt(0);  // last chunk done
#endif
          }
        }
#if !HAVE_TDM
        {  // fallback: cooperative synchronous copy of chunk c
          const uint4* s = (const uint4*)(wl + c * 16384);
          uint4* d = (uint4*)(stage + (c & 1) * 16384);
          for (int i = tid; i < 2048; i += 256) d[i] = s[i];
        }
#endif
        __syncthreads();
        const u16* buf = stage + (c & 1) * 16384;
#pragma unroll
        for (int rep = 0; rep < 2; ++rep) {
          const int tile = wave + rep * 8;
          const int ntL = tile >> 2;                 // 0..3 within chunk
          const u16* brow = buf + (ntL * 16 + nl) * 256;
          v8f acc = {};
#pragma unroll
          for (int kk = 0; kk < 8; ++kk)
            acc = wmma_bf(afr[kk], load_b_frag(brow + kk * 32, kh), acc);
          const int col = (c * 4 + ntL) * 16 + nl;
#pragma unroll
          for (int r = 0; r < 8; ++r) {
            const int t = m * 16 + kh * 8 + r;
            float v = acc[r];
            if (col < DI) xi[t * DI + col] = f2bf(v);
            else          sz[t * DI + (col - DI)] = f2bf(silu_f(v));
          }
        }
        __syncthreads();
      }
    }

    // ================= two directions =================
    for (int dir = 0; dir < 2; ++dir) {
      // ---- causal depthwise conv (K=4) + SiLU -> uB ----
      {
        const float* cw = P.cw[dir] + (size_t)(l * DI) * K_CONV;
        const float* cb = P.cb[dir] + l * DI;
        for (int dh = 0; dh < 2; ++dh) {
          const int d = tid + dh * 256;
          const float w0 = cw[d * 4 + 0], w1 = cw[d * 4 + 1];
          const float w2 = cw[d * 4 + 2], w3 = cw[d * 4 + 3];
          const float bs = cb[d];
          for (int t = 0; t < L_DIM; ++t) {
            float acc = bs;
            const float wk[4] = {w0, w1, w2, w3};
#pragma unroll
            for (int k = 0; k < K_CONV; ++k) {
              int idx = t - (K_CONV - 1) + k;
              if (idx >= 0) {
                int srcT = (dir == 0) ? idx : (L_DIM - 1 - idx);
                acc += bf2f(xi[srcT * DI + d]) * wk[k];
              }
            }
            uB[t * DI + d] = f2bf(silu_f(acc));
          }
        }
      }
      __syncthreads();

      // ---- GEMM2: p = uB(64x512) @ xpT[dir][l] (48 outs), streamed ----
      {
        const u16* wl = P.xpT + (size_t)((dir * NLAY + l) * 48) * 512;
        for (int rep = 0; rep < 2; ++rep) {
          const int tIdx = wave + rep * 8;
          if (tIdx < 12) {
            const int m = tIdx & 3, nt = tIdx >> 2;
            const u16* arow = uB + (m * 16 + nl) * DI;
            const u16* brow = wl + (nt * 16 + nl) * 512;
            v8f acc = {};
#pragma unroll
            for (int kk = 0; kk < 16; ++kk)
              acc = wmma_bf(load_a_frag(arow + kk * 32, kh),
                            load_b_frag(brow + kk * 32, kh), acc);
#pragma unroll
            for (int r = 0; r < 8; ++r)
              pR[(m * 16 + kh * 8 + r) * 48 + nt * 16 + nl] = acc[r];
          }
        }
      }
      __syncthreads();

      // ---- SSM scan: 512 channels, 2 per thread ----
      {
        for (int dh = 0; dh < 2; ++dh) {
          const int d = tid + dh * 256;
          float dtwv[R_DIM], Av[S_DIM], h[S_DIM];
#pragma unroll
          for (int r = 0; r < R_DIM; ++r)
            dtwv[r] = P.dtw[dir][(l * R_DIM + r) * DI + d];
          const float dtbv = P.dtb[dir][l * DI + d];
#pragma unroll
          for (int s = 0; s < S_DIM; ++s) {
            Av[s] = -__expf(P.Alog[dir][(size_t)(l * DI + d) * S_DIM + s]);
            h[s] = 0.f;
          }
          const float Dd = P.Dp[dir][l * DI + d];
          for (int t = 0; t < L_DIM; ++t) {
            const float* pr = pR + t * 48;
            float draw = dtbv;
#pragma unroll
            for (int r = 0; r < R_DIM; ++r) draw += pr[r] * dtwv[r];
            const float dt = softplus_f(draw);
            const float ut = bf2f(uB[t * DI + d]);
            const float du = dt * ut;
            float y = 0.f;
#pragma unroll
            for (int s = 0; s < S_DIM; ++s) {
              h[s] = __expf(dt * Av[s]) * h[s] + du * pr[16 + s];
              y += h[s] * pr[32 + s];
            }
            const float outv = y + ut * Dd;
            const int tout = (dir == 0) ? t : (L_DIM - 1 - t);
            u16* yp = yB + tout * DI + d;
            *yp = (dir == 0) ? f2bf(outv) : f2bf(bf2f(*yp) + outv);
          }
        }
      }
      __syncthreads();
    }  // dir

    // ---- gate: g = y * silu(z), in place in yB ----
    for (int i = tid; i < L_DIM * DI; i += 256)
      yB[i] = f2bf(bf2f(yB[i]) * bf2f(sz[i]));
    __syncthreads();

    // ======= GEMM3: xS += g(64x512) @ w_outT[l] (256 outs), TDM-staged ======
    // 8 chunks of 2 n-tiles (32 rows x 512 K = 32KB), double-buffered in xi.
    {
      const int m = wave & 3;
      const u16* arow = yB + (m * 16 + nl) * DI;
      v16bf afr[16];
#pragma unroll
      for (int kk = 0; kk < 16; ++kk) afr[kk] = load_a_frag(arow + kk * 32, kh);
      const u16* wl = P.w_outT + (size_t)l * 256 * 512;
      u16* stage = xi;
      if (tid < 32) tdm_load_1d(OFF_XI, wl, 4096);
      for (int c = 0; c < 8; ++c) {
        if (tid < 32) {
          if (c + 1 < 8) {
            tdm_load_1d(OFF_XI + ((c + 1) & 1) * 32768, wl + (c + 1) * 16384, 4096);
#if HAVE_TDM
            __builtin_amdgcn_s_wait_tensorcnt(1);
#endif
          } else {
#if HAVE_TDM
            __builtin_amdgcn_s_wait_tensorcnt(0);
#endif
          }
        }
#if !HAVE_TDM
        {
          const uint4* s = (const uint4*)(wl + c * 16384);
          uint4* d = (uint4*)(stage + (c & 1) * 16384);
          for (int i = tid; i < 2048; i += 256) d[i] = s[i];
        }
#endif
        __syncthreads();
        const u16* buf = stage + (c & 1) * 16384;
        const int ntL = wave >> 2;  // 0..1 within chunk
        const u16* brow = buf + (ntL * 16 + nl) * 512;
        v8f acc = {};
#pragma unroll
        for (int kk = 0; kk < 16; ++kk)
          acc = wmma_bf(afr[kk], load_b_frag(brow + kk * 32, kh), acc);
        const int col = (c * 2 + ntL) * 16 + nl;
#pragma unroll
        for (int r = 0; r < 8; ++r) {
          const int t = m * 16 + kh * 8 + r;
          u16* xp = xS + t * C_DIM + col;
          *xp = f2bf(bf2f(*xp) + acc[r]);
        }
        __syncthreads();
      }
    }
    __syncthreads();
  }  // layers

  // ================= mean over L, projection + ReLU =================
  float* mv = pR;
  {
    const int c = tid;
    float acc = 0.f;
    for (int t = 0; t < L_DIM; ++t) acc += bf2f(xS[t * C_DIM + c]);
    mv[c] = acc * (1.f / L_DIM);
  }
  __syncthreads();
  {
    float* outp = P.out + (size_t)roi * H_DIM;
    for (int hh = tid; hh < H_DIM; hh += 256) {
      float acc = P.proj_b[hh];
      for (int c = 0; c < C_DIM; ++c)
        acc += mv[c] * P.proj_w[c * H_DIM + hh];
      outp[hh] = fmaxf(acc, 0.f);
    }
  }
}

// ---------------- launcher ----------------
extern "C" void kernel_launch(void* const* d_in, const int* in_sizes, int n_in,
                              void* d_out, int out_size, void* d_ws, size_t ws_size,
                              hipStream_t stream) {
  (void)in_sizes; (void)n_in; (void)out_size; (void)ws_size;
  const float* x_flat = (const float*)d_in[0];
  const float* ln_g   = (const float*)d_in[1];
  const float* ln_b   = (const float*)d_in[2];
  const float* w_in   = (const float*)d_in[3];
  const float* w_out  = (const float*)d_in[4];
  const float* cw_f   = (const float*)d_in[5];
  const float* cb_f   = (const float*)d_in[6];
  const float* xp_f   = (const float*)d_in[7];
  const float* dtw_f  = (const float*)d_in[8];
  const float* dtb_f  = (const float*)d_in[9];
  const float* Alog_f = (const float*)d_in[10];
  const float* D_f    = (const float*)d_in[11];
  const float* cw_b   = (const float*)d_in[12];
  const float* cb_b   = (const float*)d_in[13];
  const float* xp_b   = (const float*)d_in[14];
  const float* dtw_b  = (const float*)d_in[15];
  const float* dtb_b  = (const float*)d_in[16];
  const float* Alog_b = (const float*)d_in[17];
  const float* D_b    = (const float*)d_in[18];
  const float* proj_w = (const float*)d_in[19];
  const float* proj_b = (const float*)d_in[20];

  u16* w_inT  = (u16*)((char*)d_ws + WS_WIN_OFF);
  u16* w_outT = (u16*)((char*)d_ws + WS_WOUT_OFF);
  u16* xpT    = (u16*)((char*)d_ws + WS_XP_OFF);

  {
    const int total = WS_TOTAL_ELEMS;
    prep_weights<<<(total + 255) / 256, 256, 0, stream>>>(
        w_in, w_out, xp_f, xp_b, w_inT, w_outT, xpT);
  }

  Params P;
  P.x_flat = x_flat; P.ln_g = ln_g; P.ln_b = ln_b;
  P.cw[0] = cw_f;  P.cw[1] = cw_b;
  P.cb[0] = cb_f;  P.cb[1] = cb_b;
  P.dtw[0] = dtw_f; P.dtw[1] = dtw_b;
  P.dtb[0] = dtb_f; P.dtb[1] = dtb_b;
  P.Alog[0] = Alog_f; P.Alog[1] = Alog_b;
  P.Dp[0] = D_f; P.Dp[1] = D_b;
  P.proj_w = proj_w; P.proj_b = proj_b;
  P.w_inT = w_inT; P.w_outT = w_outT; P.xpT = xpT;
  P.out = (float*)d_out;

  (void)hipFuncSetAttribute((const void*)fused_roi_mamba,
                            hipFuncAttributeMaxDynamicSharedMemorySize,
                            SMEM_BYTES);
  fused_roi_mamba<<<N_ROI, 256, SMEM_BYTES, stream>>>(P);
}
